// SlimLlamaSdpaAttention_89781996356135
// MI455X (gfx1250) — matrix operand order repaired and checked by reading, not compiled, not run on hardware
//
#include <hip/hip_runtime.h>
#include <hip/hip_bf16.h>
#include <stdint.h>

// ---------------------------------------------------------------------------
// SlimLlama SDPA block for gfx1250 (MI455X): bf16 WMMA everywhere.
//   B=2, S=2048, D=4096, 16 q-heads, 8 kv-heads, dh=128, rope theta 1e4.
// Pipeline: f32->bf16 casts -> fused QKV GEMM (wmma) -> RoPE -> flash attn
//           (async-to-LDS staging, wmma QK^T + wmma PV, online softmax)
//           -> O-proj GEMM (wmma, f32 out)
// ---------------------------------------------------------------------------

typedef __attribute__((ext_vector_type(16))) __bf16 v16bf;
typedef __attribute__((ext_vector_type(8)))  float  v8f;

#define S_LEN   2048
#define B_SZ    2
#define DMODEL  4096
#define NQH     16
#define NKVH    8
#define DH      128

__device__ __forceinline__ unsigned short f2bf(float f) {
  unsigned int u = __float_as_uint(f);
  u += 0x7FFFu + ((u >> 16) & 1u);          // round-to-nearest-even
  return (unsigned short)(u >> 16);
}
__device__ __forceinline__ float bf2f(unsigned short b) {
  return __uint_as_float(((unsigned int)b) << 16);
}

union FragBF { v16bf v; uint4 q[2]; };

// gfx1250 async global->LDS copy of 16 bytes (tracked by ASYNCcnt).
// ldsoff = LDS byte address (low 32 bits of generic shared pointer),
// gptr   = 64-bit global address.
__device__ __forceinline__ void async_copy_b128(unsigned ldsoff, const void* gptr) {
  asm volatile("global_load_async_to_lds_b128 %0, %1, off"
               :: "v"(ldsoff), "v"(gptr)
               : "memory");
}
__device__ __forceinline__ void wait_async0() {
  asm volatile("s_wait_asynccnt 0x0" ::: "memory");
}

// ------------------------------ f32 -> bf16 --------------------------------
__global__ void cvt_f32_bf16(const float* __restrict__ in,
                             unsigned short* __restrict__ out, int n) {
  int i = blockIdx.x * blockDim.x + threadIdx.x;
  if (i < n) out[i] = f2bf(in[i]);
}

// ------------------------------ RoPE (in place) ----------------------------
__global__ void rope_kernel(unsigned short* __restrict__ x,
                            const int* __restrict__ pos_ids, int H) {
  int t = blockIdx.x * blockDim.x + threadIdx.x;
  int total = B_SZ * H * S_LEN * (DH / 2);
  if (t >= total) return;
  int pair = t & 63;
  int tmp  = t >> 6;
  int s    = tmp & (S_LEN - 1);
  tmp >>= 11;
  int h = tmp % H;
  int b = tmp / H;
  int pos = pos_ids[b * S_LEN + s];
  // inv_freq = theta^(-2*pair/128) ; ln(10000) = 9.210340371976184
  float freq = (float)pos * __expf(-(float)(2 * pair) * (9.210340371976184f / 128.0f));
  float sn, cs;
  __sincosf(freq, &sn, &cs);
  size_t base = (((size_t)(b * H + h)) * S_LEN + s) * DH;
  float x1 = bf2f(x[base + pair]);
  float x2 = bf2f(x[base + 64 + pair]);
  x[base + pair]      = f2bf(x1 * cs - x2 * sn);
  x[base + 64 + pair] = f2bf(x2 * cs + x1 * sn);
}

// ------------------- bf16 WMMA GEMM: C[m,n] = sum_k A[m,k]*W[n,k] ----------
// 128 threads = 4 waves in 2(M) x 2(N); block tile 128 x 128; each wave owns
// a 64x64 sub-tile = 16 WMMA per 32-deep k-step (16 ds_read_b128 : 16 wmma).
// mode 0: write f32 C row-major (stride N)
// mode 1: QKV scatter: n<2048 -> Q[b][h][s][d]; n<3072 -> K; else V^T[b][h][d][s]
__global__ __launch_bounds__(128)
void gemm_bf16_wmma(const unsigned short* __restrict__ A,
                    const unsigned short* __restrict__ W,
                    int K, int N, int mode,
                    float* __restrict__ Cf,
                    unsigned short* __restrict__ qb,
                    unsigned short* __restrict__ kb,
                    unsigned short* __restrict__ vtb) {
  __shared__ unsigned short lA[128 * 32];
  __shared__ unsigned short lW[128 * 32];

  const int t    = threadIdx.x;
  const int lane = t & 31;
  const int wid  = t >> 5;
  const int wm   = wid >> 1;       // 0..1  -> 64 M rows
  const int wn   = wid & 1;        // 0..1  -> 64 N cols
  const int half = lane >> 4;      // 0/1
  const int l16  = lane & 15;
  const int m0   = blockIdx.y * 128;
  const int n0   = blockIdx.x * 128;

  const v8f vzero = {0.f, 0.f, 0.f, 0.f, 0.f, 0.f, 0.f, 0.f};
  v8f acc[4][4];
  for (int i = 0; i < 4; ++i)
    for (int j = 0; j < 4; ++j) acc[i][j] = vzero;

  const int chunk = t & 3;         // 4 x uint4 per 32-elem row
  const int row0  = t >> 2;        // 0..31

  for (int k0 = 0; k0 < K; k0 += 32) {
    __syncthreads();
    // stage 128x32 A-tile and 128x32 W-tile (both K-major)
    for (int p = 0; p < 4; ++p) {
      int r = row0 + p * 32;
      const uint4* ga = (const uint4*)(A + (size_t)(m0 + r) * K + k0);
      *(uint4*)&lA[r * 32 + chunk * 8] = ga[chunk];
      const uint4* gw = (const uint4*)(W + (size_t)(n0 + r) * K + k0);
      *(uint4*)&lW[r * 32 + chunk * 8] = gw[chunk];
      // speculative prefetch of next K-slab (silently dropped if OOB)
      __builtin_prefetch((const void*)(A + (size_t)(m0 + r) * K + k0 + 32 + chunk * 8), 0, 1);
      __builtin_prefetch((const void*)(W + (size_t)(n0 + r) * K + k0 + 32 + chunk * 8), 0, 1);
    }
    __syncthreads();

    // A fragments: lane l -> row m = l%16, k chunks at half*8 and 16+half*8
    FragBF af[4];
    for (int mt = 0; mt < 4; ++mt) {
      int m = wm * 64 + mt * 16 + l16;
      const unsigned short* p = &lA[m * 32 + half * 8];
      af[mt].q[0] = *(const uint4*)p;
      af[mt].q[1] = *(const uint4*)(p + 16);
    }
    // B fragments: lane l -> col n = l%16, k chunks at half*8 and 16+half*8
    for (int nt = 0; nt < 4; ++nt) {
      int n = wn * 64 + nt * 16 + l16;
      FragBF bfr;
      const unsigned short* p = &lW[n * 32 + half * 8];
      bfr.q[0] = *(const uint4*)p;
      bfr.q[1] = *(const uint4*)(p + 16);
      for (int mt = 0; mt < 4; ++mt)
        acc[mt][nt] = __builtin_amdgcn_wmma_f32_16x16x32_bf16(
            false, af[mt].v, false, bfr.v, (short)0, acc[mt][nt], false, false);
    }
  }

  // epilogue: C layout -> lane l holds (M = half*8 + r, N = l%16) per tile
  for (int mt = 0; mt < 4; ++mt)
    for (int nt = 0; nt < 4; ++nt) {
      int col = n0 + wn * 64 + nt * 16 + l16;
      for (int r = 0; r < 8; ++r) {
        int Mg = m0 + wm * 64 + mt * 16 + half * 8 + r;
        float v = acc[mt][nt][r];
        if (mode == 0) {
          Cf[(size_t)Mg * N + col] = v;
        } else {
          int b = Mg / S_LEN, s = Mg % S_LEN;
          unsigned short bv = f2bf(v);
          if (col < 2048) {
            int hh = col >> 7, d = col & 127;
            qb[((size_t)(b * NQH + hh) * S_LEN + s) * DH + d] = bv;
          } else if (col < 3072) {
            int c2 = col - 2048, hh = c2 >> 7, d = c2 & 127;
            kb[((size_t)(b * NKVH + hh) * S_LEN + s) * DH + d] = bv;
          } else {
            int c2 = col - 3072, hh = c2 >> 7, d = c2 & 127;
            vtb[((size_t)(b * NKVH + hh) * DH + d) * S_LEN + s] = bv;
          }
        }
      }
    }
}

// ----------------------- flash attention (bf16 WMMA) -----------------------
// grid (S/64, NQH, B); 128 threads = 4 waves; wave w owns q rows q0+16w..+15.
// K/V tiles staged with gfx1250 async global->LDS copies (ASYNCcnt).
__global__ __launch_bounds__(128)
void flash_attn_wmma(const unsigned short* __restrict__ qb,
                     const unsigned short* __restrict__ kb,
                     const unsigned short* __restrict__ vtb,
                     unsigned short* __restrict__ attn) {
  __shared__ unsigned short lK[64 * 128];   // [kv_local][dh]
  __shared__ unsigned short lV[128 * 64];   // [dh][kv_local]
  __shared__ unsigned short lP[4 * 16 * 64];// per-wave P tiles [m][kv_local]

  const int t    = threadIdx.x;
  const int lane = t & 31;
  const int w    = t >> 5;
  const int half = lane >> 4, l16 = lane & 15;
  const int qblk = blockIdx.x;
  const int h    = blockIdx.y;
  const int b    = blockIdx.z;
  const int kvh  = h >> 1;                  // GQA groups = 2
  const int q0   = qblk * 64;

  // preload Q fragments (4 k-steps over dh=128), straight from global
  FragBF qf[4];
  {
    int m = q0 + w * 16 + l16;
    const unsigned short* qp = qb + ((size_t)(b * NQH + h) * S_LEN + m) * DH;
    for (int kt = 0; kt < 4; ++kt) {
      qf[kt].q[0] = *(const uint4*)(qp + kt * 32 + half * 8);
      qf[kt].q[1] = *(const uint4*)(qp + kt * 32 + 16 + half * 8);
    }
  }

  const v8f vzero = {0.f, 0.f, 0.f, 0.f, 0.f, 0.f, 0.f, 0.f};
  v8f accO[8];
  for (int i = 0; i < 8; ++i) accO[i] = vzero;
  float m_i[8], l_i[8];
  for (int r = 0; r < 8; ++r) { m_i[r] = -3.0e38f; l_i[r] = 0.f; }
  const float scale = 0.08838834764831845f;  // 1/sqrt(128)

  for (int j = 0; j <= qblk; ++j) {
    __syncthreads();                          // prior iter done reading lK/lV
    { // stage K tile: 64 kv rows x 128 dh  (async, 8 x b128 per thread)
      int c = t & 15, r0 = t >> 4;
      for (int p = 0; p < 8; ++p) {
        int r = r0 + p * 8;
        const unsigned short* gp =
            kb + ((size_t)(b * NKVH + kvh) * S_LEN + (j * 64 + r)) * DH + c * 8;
        unsigned la = (unsigned)(uintptr_t)&lK[r * 128 + c * 8];
        async_copy_b128(la, gp);
      }
    }
    { // stage V^T tile: 128 dh rows x 64 kv (async, 8 x b128 per thread)
      int c = t & 7, d0 = t >> 3;
      for (int p = 0; p < 8; ++p) {
        int d = d0 + p * 16;
        const unsigned short* gp =
            vtb + ((size_t)(b * NKVH + kvh) * DH + d) * S_LEN + j * 64 + c * 8;
        unsigned la = (unsigned)(uintptr_t)&lV[d * 64 + c * 8];
        async_copy_b128(la, gp);
      }
    }
    wait_async0();                            // ASYNCcnt == 0: tiles landed
    __syncthreads();

    // scores: 4 N-tiles of 16 kv cols, contract dh over 4 WMMA each
    v8f sc[4];
    for (int nt = 0; nt < 4; ++nt) {
      v8f a = vzero;
      int n = nt * 16 + l16;
      for (int kt = 0; kt < 4; ++kt) {
        FragBF kfr;
        const unsigned short* p = &lK[n * 128 + kt * 32 + half * 8];
        kfr.q[0] = *(const uint4*)p;
        kfr.q[1] = *(const uint4*)(p + 16);
        a = __builtin_amdgcn_wmma_f32_16x16x32_bf16(
            false, qf[kt].v, false, kfr.v, (short)0, a, false, false);
      }
      sc[nt] = a;
    }

    // scale + causal mask + online softmax
    float mnew[8];
    for (int r = 0; r < 8; ++r) mnew[r] = m_i[r];
    for (int nt = 0; nt < 4; ++nt) {
      int col = j * 64 + nt * 16 + l16;
      for (int r = 0; r < 8; ++r) {
        int rowg = q0 + w * 16 + half * 8 + r;
        float v = sc[nt][r] * scale;
        if (col > rowg) v = -3.0e38f;
        sc[nt][r] = v;
        mnew[r] = fmaxf(mnew[r], v);
      }
    }
    for (int r = 0; r < 8; ++r)
      for (int off = 1; off < 16; off <<= 1)
        mnew[r] = fmaxf(mnew[r], __shfl_xor(mnew[r], off, 32));
    float rs[8];
    for (int r = 0; r < 8; ++r) {
      float alpha = __expf(m_i[r] - mnew[r]);
      m_i[r] = mnew[r];
      l_i[r] *= alpha;
      rs[r] = 0.f;
      for (int nt = 0; nt < 8; ++nt) accO[nt][r] *= alpha;
    }
    for (int nt = 0; nt < 4; ++nt)
      for (int r = 0; r < 8; ++r) {
        float pv = __expf(sc[nt][r] - m_i[r]);
        rs[r] += pv;
        lP[w * 1024 + (half * 8 + r) * 64 + nt * 16 + l16] = f2bf(pv);
      }
    for (int r = 0; r < 8; ++r) {
      for (int off = 1; off < 16; off <<= 1) rs[r] += __shfl_xor(rs[r], off, 32);
      l_i[r] += rs[r];
    }
    __syncthreads();                          // P visible (C-layout -> A-layout)

    // PV: contract 64 kv in 2 k-steps; 8 dh N-tiles
    for (int kt = 0; kt < 2; ++kt) {
      FragBF pf;
      const unsigned short* pp = &lP[w * 1024 + l16 * 64 + kt * 32 + half * 8];
      pf.q[0] = *(const uint4*)pp;
      pf.q[1] = *(const uint4*)(pp + 16);
      for (int nt = 0; nt < 8; ++nt) {
        FragBF vf;
        const unsigned short* vp = &lV[(nt * 16 + l16) * 64 + kt * 32 + half * 8];
        vf.q[0] = *(const uint4*)vp;
        vf.q[1] = *(const uint4*)(vp + 16);
        accO[nt] = __builtin_amdgcn_wmma_f32_16x16x32_bf16(
            false, pf.v, false, vf.v, (short)0, accO[nt], false, false);
      }
    }
  }

  // normalize + store attn (bf16, [b*S + s][h*128 + d]) for the O-projection
  for (int nt = 0; nt < 8; ++nt) {
    int col = h * DH + nt * 16 + l16;
    for (int r = 0; r < 8; ++r) {
      int rowg = q0 + w * 16 + half * 8 + r;
      float v = accO[nt][r] / l_i[r];
      attn[((size_t)(b * S_LEN) + rowg) * (NQH * DH) + col] = f2bf(v);
    }
  }
}

// ------------------------------- launcher ----------------------------------
extern "C" void kernel_launch(void* const* d_in, const int* in_sizes, int n_in,
                              void* d_out, int out_size, void* d_ws, size_t ws_size,
                              hipStream_t stream) {
  const float* hs  = (const float*)d_in[0];
  const int*   pos = (const int*)d_in[1];
  const float* wq  = (const float*)d_in[2];
  const float* wk  = (const float*)d_in[3];
  const float* wv  = (const float*)d_in[4];
  const float* wo  = (const float*)d_in[5];
  float* out = (float*)d_out;

  unsigned short* ws      = (unsigned short*)d_ws;
  unsigned short* hs_bf   = ws;                            // 16,777,216
  unsigned short* wqkv_bf = hs_bf   + (size_t)16777216;    // 16,777,216 (wq|wk|wv)
  unsigned short* wo_bf   = wqkv_bf + (size_t)16777216;    //  8,388,608
  unsigned short* q_bf    = wo_bf   + (size_t)8388608;     //  8,388,608
  unsigned short* k_bf    = q_bf    + (size_t)8388608;     //  4,194,304
  unsigned short* vt_bf   = k_bf    + (size_t)4194304;     //  4,194,304 (V transposed)
  unsigned short* attn_bf = vt_bf   + (size_t)4194304;     //  8,388,608

  cvt_f32_bf16<<<16777216 / 256, 256, 0, stream>>>(hs, hs_bf, 16777216);
  cvt_f32_bf16<<< 8388608 / 256, 256, 0, stream>>>(wq, wqkv_bf, 8388608);
  cvt_f32_bf16<<< 4194304 / 256, 256, 0, stream>>>(wk, wqkv_bf + 8388608, 4194304);
  cvt_f32_bf16<<< 4194304 / 256, 256, 0, stream>>>(wv, wqkv_bf + 12582912, 4194304);
  cvt_f32_bf16<<< 8388608 / 256, 256, 0, stream>>>(wo, wo_bf, 8388608);

  // fused QKV projection: M=B*S=4096, N=4096 (2048 q | 1024 k | 1024 v), K=4096
  gemm_bf16_wmma<<<dim3(32, 32), 128, 0, stream>>>(
      hs_bf, wqkv_bf, 4096, 4096, 1, nullptr, q_bf, k_bf, vt_bf);

  rope_kernel<<<(B_SZ * NQH  * S_LEN * 64) / 256, 256, 0, stream>>>(q_bf, pos, NQH);
  rope_kernel<<<(B_SZ * NKVH * S_LEN * 64) / 256, 256, 0, stream>>>(k_bf, pos, NKVH);

  flash_attn_wmma<<<dim3(S_LEN / 64, NQH, B_SZ), 128, 0, stream>>>(
      q_bf, k_bf, vt_bf, attn_bf);

  // output projection: M=4096, N=4096, K=2048, f32 result straight to d_out
  gemm_bf16_wmma<<<dim3(32, 32), 128, 0, stream>>>(
      attn_bf, wo_bf, 2048, 4096, 0, out, nullptr, nullptr, nullptr);
}
